// Router_28767690949184
// MI455X (gfx1250) — matrix operand order repaired and checked by reading, not compiled, not run on hardware
//
#include <hip/hip_runtime.h>
#include <cstdint>
#include <cstddef>

typedef __attribute__((ext_vector_type(16))) __bf16 v16bf;
typedef __attribute__((ext_vector_type(8)))  __bf16 v8bf;
typedef __attribute__((ext_vector_type(8)))  float  v8f;

#define N_SAMPLES 16384
#define C_CH 64
#define E_EXP 16
#define D_DIM 1344
#define TILE 16
#define TILES_PER_BLOCK 4
#define SAMPLES_PER_BLOCK (TILE * TILES_PER_BLOCK)
#define BLOCK_THREADS 128
#define GRID_MAIN (N_SAMPLES / SAMPLES_PER_BLOCK)   // 256

#define EMB_STRIDE 1352     // bf16 elements, padded (1344+8) for bank spread
#define KEYS_STRIDE 1352
#define ACC_STRIDE 1348     // f32 elements, padded (1344+4)

// ---- dynamic LDS layout (bytes) ----
#define OFF_KEYS   0
#define SZ_KEYS    (E_EXP * KEYS_STRIDE * 2)        // 43264
#define OFF_EMB    (OFF_KEYS + SZ_KEYS)
#define SZ_EMB     (TILE * EMB_STRIDE * 2)          // 43264
#define OFF_ACC    (OFF_EMB + SZ_EMB)
#define SZ_ACC     (E_EXP * ACC_STRIDE * 4)         // 86272
#define OFF_LOG    (OFF_ACC + SZ_ACC)
#define SZ_LOG     (TILE * E_EXP * 4)               // 1024
#define OFF_WSCT   (OFF_LOG + SZ_LOG)
#define SZ_WSCT    (E_EXP * TILE * 2)               // 512
#define OFF_NRM    (OFF_WSCT + SZ_WSCT)
#define SZ_NRM     (TILE * 4)
#define OFF_SUM    (OFF_NRM + SZ_NRM)
#define SZ_SUM     (E_EXP * 4)
#define SMEM_TOTAL (OFF_SUM + SZ_SUM)               // 174464 B < 320 KB WGP LDS

__device__ __forceinline__ float4 f4max(float4 a, float4 b) {
    return make_float4(fmaxf(a.x, b.x), fmaxf(a.y, b.y),
                       fmaxf(a.z, b.z), fmaxf(a.w, b.w));
}

// ---------------------------------------------------------------------------
// Kernel 0: zero the global centroid accumulator (new_keys region of d_out)
// and the per-expert weight sums in d_ws.
// ---------------------------------------------------------------------------
__global__ void router_init(float* __restrict__ out, float* __restrict__ ws_sums) {
    int g = blockIdx.x * blockDim.x + threadIdx.x;
    if (g < E_EXP * D_DIM) out[(size_t)N_SAMPLES * E_EXP + g] = 0.0f;
    if (g < E_EXP) ws_sums[g] = 0.0f;
}

// ---------------------------------------------------------------------------
// Kernel 1: fused SPP-embedding + WMMA routing + WMMA centroid accumulation.
// 256 blocks x 128 threads (4 waves). 64 samples per block, 16-sample tiles.
// ---------------------------------------------------------------------------
__global__ __launch_bounds__(BLOCK_THREADS, 1)
void router_main(const float* __restrict__ patch,
                 const float* __restrict__ keys,
                 const float* __restrict__ thr_p,
                 float* __restrict__ out,        // [N*E wf][E*D acc]
                 float* __restrict__ ws_sums) {
    extern __shared__ unsigned char smem[];
    __bf16* keys_bf = (__bf16*)(smem + OFF_KEYS);   // [E][KEYS_STRIDE]
    __bf16* emb_bf  = (__bf16*)(smem + OFF_EMB);    // [TILE][EMB_STRIDE]
    float*  accS    = (float*) (smem + OFF_ACC);    // [E][ACC_STRIDE]
    float*  logitsS = (float*) (smem + OFF_LOG);    // [TILE][E]
    __bf16* wscT    = (__bf16*)(smem + OFF_WSCT);   // [E][TILE]
    float*  nrm     = (float*) (smem + OFF_NRM);    // [TILE]
    float*  sumsL   = (float*) (smem + OFF_SUM);    // [E]

    const int tid  = threadIdx.x;
    const int lane = tid & 31;
    const int wv   = tid >> 5;
    const int row  = lane & 15;
    const int hl8  = (lane < 16) ? 0 : 8;   // K half-select for A operand

    // ---- block init: stage keys as bf16, zero centroid acc + sums ----
    for (int i = tid; i < E_EXP * D_DIM; i += BLOCK_THREADS) {
        int e = i / D_DIM, d = i - e * D_DIM;
        keys_bf[e * KEYS_STRIDE + d] = (__bf16)keys[i];
    }
    for (int i = tid; i < E_EXP * ACC_STRIDE; i += BLOCK_THREADS) accS[i] = 0.0f;
    if (tid < E_EXP) sumsL[tid] = 0.0f;
    const float thrv = *thr_p;
    __syncthreads();

    for (int t = 0; t < TILES_PER_BLOCK; ++t) {
        const int m0 = blockIdx.x * SAMPLES_PER_BLOCK + t * TILE;

        if (tid < TILE) nrm[tid] = 0.0f;
        for (int i = tid; i < TILE * E_EXP; i += BLOCK_THREADS) logitsS[i] = 0.0f;
        __syncthreads();

        // ---- Phase 1: SPP max-pool embedding (16 samples x 64 channels) ----
        #pragma unroll
        for (int j = 0; j < 8; ++j) {
            const int p = j * BLOCK_THREADS + tid;       // (m,c) pair
            const int m = p >> 6, c = p & 63;
            const float4* src = reinterpret_cast<const float4*>(patch)
                                + ((size_t)(m0 + m) * C_CH + c) * 16;
            float4 ch[16];
            #pragma unroll
            for (int k = 0; k < 16; ++k) ch[k] = src[k];

            float q[4][4];                               // 2x2-block maxes (s=4)
            #pragma unroll
            for (int i2 = 0; i2 < 4; ++i2) {
                float4 ma = f4max(ch[4 * i2 + 0], ch[4 * i2 + 2]);
                float4 mb = f4max(ch[4 * i2 + 1], ch[4 * i2 + 3]);
                q[i2][0] = fmaxf(ma.x, ma.y); q[i2][1] = fmaxf(ma.z, ma.w);
                q[i2][2] = fmaxf(mb.x, mb.y); q[i2][3] = fmaxf(mb.z, mb.w);
            }
            float p2[2][2];                              // 4x4-block maxes (s=2)
            #pragma unroll
            for (int a = 0; a < 2; ++a)
                #pragma unroll
                for (int b = 0; b < 2; ++b)
                    p2[a][b] = fmaxf(fmaxf(q[2*a][2*b],   q[2*a][2*b+1]),
                                     fmaxf(q[2*a+1][2*b], q[2*a+1][2*b+1]));
            float p1 = fmaxf(fmaxf(p2[0][0], p2[0][1]), fmaxf(p2[1][0], p2[1][1]));

            __bf16* er = emb_bf + m * EMB_STRIDE;
            float ss = p1 * p1;
            er[c] = (__bf16)p1;                                       // level 1
            #pragma unroll
            for (int a = 0; a < 2; ++a)
                #pragma unroll
                for (int b = 0; b < 2; ++b) {
                    er[64 + c * 4 + a * 2 + b] = (__bf16)p2[a][b];    // level 2
                    ss += p2[a][b] * p2[a][b];
                }
            #pragma unroll
            for (int i2 = 0; i2 < 4; ++i2)
                #pragma unroll
                for (int j2 = 0; j2 < 4; ++j2) {
                    er[320 + c * 16 + i2 * 4 + j2] = (__bf16)q[i2][j2]; // level 4
                    ss += q[i2][j2] * q[i2][j2];
                }
            __hip_atomic_fetch_add(&nrm[m], ss, __ATOMIC_RELAXED,
                                   __HIP_MEMORY_SCOPE_WORKGROUP);
        }
        __syncthreads();

        // ---- Phase 2: logits = emb @ keys^T via v_wmma_f32_16x16x32_bf16 ----
        // K = 1344 = 42 chunks of 32, strided over the 4 waves.
        v8f creg;
        #pragma unroll
        for (int r = 0; r < 8; ++r) creg[r] = 0.0f;
        for (int chunk = wv; chunk < 42; chunk += 4) {
            const int k0 = chunk * 32;
            // A (16x32, M=sample): lanes<16 K=[0..7],[16..23]; lanes>=16 +8
            const v8bf* ap = (const v8bf*)(emb_bf + row * EMB_STRIDE + k0 + hl8);
            v8bf alo = ap[0], ahi = ap[2];               // +16 elements = +2 v8bf
            // B (32x16, N=expert): lanes<16 K=[0..7],[16..23]; lanes>=16 +8
            const v8bf* bp = (const v8bf*)(keys_bf + row * KEYS_STRIDE + k0
                                           + hl8);
            v8bf blo = bp[0], bhi = bp[2];
            v16bf av, bv;
            #pragma unroll
            for (int i = 0; i < 8; ++i) {
                av[i] = alo[i]; av[i + 8] = ahi[i];
                bv[i] = blo[i]; bv[i + 8] = bhi[i];
            }
            creg = __builtin_amdgcn_wmma_f32_16x16x32_bf16(
                       false, av, false, bv, (short)0, creg, false, false);
        }
        #pragma unroll
        for (int r = 0; r < 8; ++r)
            __hip_atomic_fetch_add(&logitsS[(r + hl8) * E_EXP + row], creg[r],
                                   __ATOMIC_RELAXED, __HIP_MEMORY_SCOPE_WORKGROUP);
        __syncthreads();

        // ---- Phase 3: softmax + adaptive-threshold gating (1 thread/sample) ----
        if (tid < TILE) {
            const float inv = 1.0f / fmaxf(sqrtf(nrm[tid]), 1e-12f);
            float w[E_EXP], mx = -1e30f;
            #pragma unroll
            for (int e = 0; e < E_EXP; ++e) {
                w[e] = logitsS[tid * E_EXP + e] * inv;
                mx = fmaxf(mx, w[e]);
            }
            float s = 0.0f;
            #pragma unroll
            for (int e = 0; e < E_EXP; ++e) { w[e] = __expf(w[e] - mx); s += w[e]; }
            const float invs = 1.0f / s;
            float maxw = 0.0f;
            #pragma unroll
            for (int e = 0; e < E_EXP; ++e) { w[e] *= invs; maxw = fmaxf(maxw, w[e]); }
            const float at = fminf(fmaxf(thrv * (2.0f - maxw), 0.01f), 0.8f);
            float wsum = 0.0f;
            #pragma unroll
            for (int e = 0; e < E_EXP; ++e) {
                const float sg = 1.0f / (1.0f + __expf(-10.0f * (w[e] - at)));
                w[e] *= sg; wsum += w[e];
            }
            const float invw = 1.0f / fmaxf(wsum, 1e-8f);
            const size_t gm = (size_t)(m0 + tid) * E_EXP;
            #pragma unroll
            for (int e = 0; e < E_EXP; ++e) {
                const float wf = w[e] * invw;
                out[gm + e] = wf;
                wscT[e * TILE + tid] = (__bf16)(wf * inv);  // fold 1/||emb||
                __hip_atomic_fetch_add(&sumsL[e], wf, __ATOMIC_RELAXED,
                                       __HIP_MEMORY_SCOPE_WORKGROUP);
            }
        }
        __syncthreads();

        // ---- Phase 4: centroid acc += wsc^T @ emb via WMMA (K=16, zero-pad 32) ----
        v16bf av2;
        {
            const v8bf* wp = (const v8bf*)(wscT + row * TILE + hl8);
            v8bf w8 = *wp;
            #pragma unroll
            for (int i = 0; i < 8; ++i) { av2[i] = w8[i]; av2[i + 8] = (__bf16)0.0f; }
        }
        for (int dt = wv; dt < D_DIM / 16; dt += 4) {    // 84 D-tiles / 4 waves
            const int n0 = dt * 16;
            v16bf bv2;
            {
                #pragma unroll
                for (int i = 0; i < 8; ++i)
                    bv2[i] = emb_bf[(i + hl8) * EMB_STRIDE + n0 + row];
                #pragma unroll
                for (int i = 8; i < 16; ++i) bv2[i] = (__bf16)0.0f;  // K>=16 pad
            }
            v8f cc;
            #pragma unroll
            for (int r = 0; r < 8; ++r) cc[r] = 0.0f;
            cc = __builtin_amdgcn_wmma_f32_16x16x32_bf16(
                     false, av2, false, bv2, (short)0, cc, false, false);
            #pragma unroll
            for (int r = 0; r < 8; ++r)
                accS[(r + hl8) * ACC_STRIDE + n0 + row] += cc[r];    // wave-owned
        }
        __syncthreads();
    }

    // ---- block flush: one global f32-atomic pass into new_keys acc region ----
    float* accG = out + (size_t)N_SAMPLES * E_EXP;
    for (int i = tid; i < E_EXP * D_DIM; i += BLOCK_THREADS) {
        int e = i / D_DIM, d = i - e * D_DIM;
        __hip_atomic_fetch_add(&accG[i], accS[e * ACC_STRIDE + d],
                               __ATOMIC_RELAXED, __HIP_MEMORY_SCOPE_AGENT);
    }
    if (tid < E_EXP)
        __hip_atomic_fetch_add(&ws_sums[tid], sumsL[tid],
                               __ATOMIC_RELAXED, __HIP_MEMORY_SCOPE_AGENT);
}

// ---------------------------------------------------------------------------
// Kernel 2: finalize EMA key update (one block per expert).
// ---------------------------------------------------------------------------
__device__ __forceinline__ float blockReduceSum(float v, float* red) {
    const int t = threadIdx.x;
    red[t] = v;
    __syncthreads();
    for (int s = 64; s > 0; s >>= 1) {
        if (t < s) red[t] += red[t + s];
        __syncthreads();
    }
    float r = red[0];
    __syncthreads();
    return r;
}

__global__ void router_finalize(const float* __restrict__ keys,
                                const float* __restrict__ ws_sums,
                                float* __restrict__ out) {
    __shared__ float red[128];
    const int e = blockIdx.x, t = threadIdx.x;
    const float s    = ws_sums[e];
    const float cinv = 1.0f / (s + 1e-8f);
    float* rowp = out + (size_t)N_SAMPLES * E_EXP + (size_t)e * D_DIM;

    float cv[11], kv[11];
    int cnt = 0;
    float ss = 0.0f;
    for (int d = t; d < D_DIM; d += 128) {
        const float c = rowp[d] * cinv;
        cv[cnt] = c;
        kv[cnt] = keys[e * D_DIM + d];
        ss += c * c;
        ++cnt;
    }
    const float n1 = sqrtf(blockReduceSum(ss, red));
    const float i1 = 1.0f / fmaxf(n1, 1e-12f);

    float ss2 = 0.0f;
    for (int i = 0; i < cnt; ++i) {
        const float u = 0.99f * kv[i] + 0.01f * cv[i] * i1;
        cv[i] = u;
        ss2 += u * u;
    }
    const float n2 = sqrtf(blockReduceSum(ss2, red));
    const float i2 = 1.0f / fmaxf(n2, 1e-12f);

    int idx = 0;
    for (int d = t; d < D_DIM; d += 128) {
        rowp[d] = (s > 0.0f) ? cv[idx] * i2 : kv[idx];
        ++idx;
    }
}

// ---------------------------------------------------------------------------
extern "C" void kernel_launch(void* const* d_in, const int* in_sizes, int n_in,
                              void* d_out, int out_size, void* d_ws, size_t ws_size,
                              hipStream_t stream) {
    (void)in_sizes; (void)n_in; (void)out_size; (void)ws_size;
    const float* patch = (const float*)d_in[0];
    const float* keys  = (const float*)d_in[1];
    const float* thr   = (const float*)d_in[2];
    float* out     = (float*)d_out;
    float* ws_sums = (float*)d_ws;   // 16 floats

    router_init<<<(E_EXP * D_DIM + 127) / 128, 128, 0, stream>>>(out, ws_sums);
    router_main<<<GRID_MAIN, BLOCK_THREADS, SMEM_TOTAL, stream>>>(
        patch, keys, thr, out, ws_sums);
    router_finalize<<<E_EXP, 128, 0, stream>>>(keys, ws_sums, out);
}
